// ProbAttention_77489799954814
// MI455X (gfx1250) — matrix operand order repaired and checked by reading, not compiled, not run on hardware
//
#include <hip/hip_runtime.h>
#include <stdint.h>

#define NB 16
#define NL 2048
#define NS 2048
#define NE 512

// LDS tile geometry (bytes). K tile: 32 rows x 1024B data, padded to 1040B.
// V^T tile: 512 rows x 64B data, padded to 80B. Double buffered.
#define KROWB 1040
#define VROWB 80
#define KBYTES (32 * KROWB)     // 33280
#define VBYTES (512 * VROWB)    // 40960
#define BUFB   (KBYTES + VBYTES)// 74240
#define SMEMB  (2 * BUFB)       // 148480

typedef __attribute__((ext_vector_type(16))) __bf16 v16bf;
typedef __attribute__((ext_vector_type(8)))  float  v8f;

union V16BF {
    v16bf v;
    uint4 u4[2];
    uint32_t u[8];
};

__device__ __forceinline__ uint32_t bf16_rne(float x) {
    uint32_t b = __float_as_uint(x);
    b += 0x7FFFu + ((b >> 16) & 1u);
    return b >> 16;
}
__device__ __forceinline__ uint32_t pack2(float lo, float hi) {
    return bf16_rne(lo) | (bf16_rne(hi) << 16);
}

// CDNA5 async global->LDS copy (ASYNCcnt tracked); per-lane 16B transfer.
__device__ __forceinline__ void async_load_b128(uint32_t lds_off, const void* gptr) {
    asm volatile("global_load_async_to_lds_b128 %0, %1, off"
                 :: "v"(lds_off), "v"((uint64_t)(uintptr_t)gptr)
                 : "memory");
}

// ---------------- pre-pass 1: K fp32 -> bf16 (row-major [B,S,E]) -------------
__global__ __launch_bounds__(256)
void convert_k_kernel(const float* __restrict__ K, uint16_t* __restrict__ Kb) {
    size_t i = ((size_t)blockIdx.x * blockDim.x + threadIdx.x) * 8;
    const float4* src = (const float4*)(K + i);
    float4 a = src[0], b = src[1];
    uint4 out;
    out.x = pack2(a.x, a.y);
    out.y = pack2(a.z, a.w);
    out.z = pack2(b.x, b.y);
    out.w = pack2(b.z, b.w);
    *(uint4*)(Kb + i) = out;
}

// ---------------- pre-pass 2: V fp32 [B,S,E] -> V^T bf16 [B,E,S] -------------
__global__ __launch_bounds__(256)
void transpose_v_kernel(const float* __restrict__ V, uint16_t* __restrict__ VT) {
    __shared__ float tile[32][33];
    const int tx = threadIdx.x;
    const int ty = threadIdx.y;
    const int s0 = blockIdx.x * 32;
    const int e0 = blockIdx.y * 32;
    const int b  = blockIdx.z;
    const float* vb = V + (size_t)b * NS * NE;
#pragma unroll
    for (int i = 0; i < 4; ++i) {
        int s = s0 + ty + i * 8;
        tile[ty + i * 8][tx] = vb[(size_t)s * NE + (e0 + tx)];
    }
    __syncthreads();
    uint16_t* vt = VT + (size_t)b * NE * NS;
#pragma unroll
    for (int i = 0; i < 4; ++i) {
        int e = e0 + ty + i * 8;
        vt[(size_t)e * NS + (s0 + tx)] = (uint16_t)bf16_rne(tile[tx][ty + i * 8]);
    }
}

// Cooperative fill of one double-buffer slot: K tile (32 keys x 512e bf16) and
// V^T tile (512e x 32 keys bf16) via async copies. 32 x b128 per lane.
__device__ __forceinline__ void fill_tiles(uint32_t ldsK, uint32_t ldsV,
                                           const uint16_t* Kg, const uint16_t* VTg,
                                           int k0, int tid) {
#pragma unroll
    for (int t = 0; t < 16; ++t) {
        int u   = t * 128 + tid;      // 0..2047 16B units
        int key = u >> 6;             // 64 units per 1KB key row
        int cb  = (u & 63) * 16;
        async_load_b128(ldsK + key * KROWB + cb,
                        (const char*)Kg + (size_t)(k0 + key) * (NE * 2) + cb);
    }
#pragma unroll
    for (int t = 0; t < 16; ++t) {
        int u  = t * 128 + tid;       // 0..2047 16B units
        int e  = u >> 2;              // 4 units per 64B e-row
        int cb = (u & 3) * 16;
        async_load_b128(ldsV + e * VROWB + cb,
                        (const char*)VTg + ((size_t)e * NS + k0) * 2 + cb);
    }
}

// ---------------- main: flash attention ----------------
// 4 waves/WG, 64 queries/WG. K/V tiles double-buffered in LDS via async DMA,
// shared by all 4 waves (4x L2 traffic reduction). Each wave: S^T = K*Q^T via
// v_wmma (lane = query -> lane-local online softmax), then O += P*V.
__global__ __launch_bounds__(128)
void flash_attn_kernel(const float* __restrict__ Q,
                       const uint16_t* __restrict__ Kb,
                       const uint16_t* __restrict__ VT,
                       float* __restrict__ O) {
    extern __shared__ char smem[];
    const int tid  = threadIdx.x;
    const int lane = tid & 31;
    const int wave = tid >> 5;
    const int m = lane & 15;          // query col (C-layout N) for this lane
    const int h = lane >> 4;
    const int q0wg = (gridDim.x - 1 - blockIdx.x) * 64;   // big blocks first
    const int q0 = q0wg + wave * 16;
    const int b  = blockIdx.y;
    const float scale = 0.04419417382415922f;   // 1/sqrt(512)

    const uint16_t* Kg  = Kb + (size_t)b * NS * NE;
    const uint16_t* VTg = VT + (size_t)b * NE * NS;
    const uint32_t smembase = (uint32_t)(uintptr_t)smem;  // LDS byte offset

    // ---- Q into B-operand registers: qb[c] covers e = c*32..c*32+31 ----
    v16bf qb[16];
    {
        const float* qrow = Q + ((size_t)b * NL + (q0 + m)) * NE;
#pragma unroll
        for (int c = 0; c < 16; ++c) {
            const float4* p = (const float4*)(qrow + c * 32 + h * 16);
            float4 f0 = p[0], f1 = p[1], f2 = p[2], f3 = p[3];
            V16BF t;
            t.u[0] = pack2(f0.x, f0.y); t.u[1] = pack2(f0.z, f0.w);
            t.u[2] = pack2(f1.x, f1.y); t.u[3] = pack2(f1.z, f1.w);
            t.u[4] = pack2(f2.x, f2.y); t.u[5] = pack2(f2.z, f2.w);
            t.u[6] = pack2(f3.x, f3.y); t.u[7] = pack2(f3.z, f3.w);
            qb[c] = t.v;
        }
    }

    v8f o[32];
#pragma unroll
    for (int e = 0; e < 32; ++e) o[e] = v8f{};

    float mrow = -INFINITY;
    float lsum = 0.0f;

    const int nsteps = (q0wg + 64 + 31) >> 5;   // whole-WG causal extent

    // prologue: fill buffer 0 for step 0
    fill_tiles(smembase, smembase + KBYTES, Kg, VTg, 0, tid);

#pragma unroll 1
    for (int i = 0; i < nsteps; ++i) {
        const int k0 = i * 32;
        // prefetch next step into the other buffer, then wait for this step's
        // batch (async completes in-order: <=32 outstanding => batch i done)
        if (i + 1 < nsteps) {
            const uint32_t nb = smembase + ((i + 1) & 1) * BUFB;
            fill_tiles(nb, nb + KBYTES, Kg, VTg, (i + 1) * 32, tid);
            asm volatile("s_wait_asynccnt 32" ::: "memory");
        } else {
            asm volatile("s_wait_asynccnt 0" ::: "memory");
        }
        __syncthreads();   // all waves' fills for step i visible

        const char* kbuf = smem + (size_t)(i & 1) * BUFB;
        const char* vbuf = kbuf + KBYTES;

        // ---- S^T tiles (keys x queries) from LDS K tile + reg Q ----
        v8f st[2];
        st[0] = v8f{}; st[1] = v8f{};
#pragma unroll
        for (int t = 0; t < 2; ++t) {
            const char* krow = kbuf + (size_t)(t * 16 + m) * KROWB + h * 32;
#pragma unroll
            for (int c = 0; c < 16; ++c) {
                V16BF a;
                a.u4[0] = *(const uint4*)(krow + c * 64);
                a.u4[1] = *(const uint4*)(krow + c * 64 + 16);
                st[t] = __builtin_amdgcn_wmma_f32_16x16x32_bf16(
                    false, a.v, false, qb[c], (short)0, st[t], false, false);
            }
        }

        // ---- scale + causal mask + online softmax (lane-local) ----
        float p[2][8];
        float tmax = -INFINITY;
#pragma unroll
        for (int t = 0; t < 2; ++t)
#pragma unroll
            for (int r = 0; r < 8; ++r) {
                int key = k0 + t * 16 + r + 8 * h;
                float s = st[t][r] * scale;
                s = (key > q0 + m) ? -INFINITY : s;
                p[t][r] = s;
                tmax = fmaxf(tmax, s);
            }
        tmax = fmaxf(tmax, __shfl_xor(tmax, 16, 32));
        float mnew  = fmaxf(mrow, tmax);
        float alpha = __expf(mrow - mnew);
        float rsum = 0.0f;
#pragma unroll
        for (int t = 0; t < 2; ++t)
#pragma unroll
            for (int r = 0; r < 8; ++r) {
                float ev = __expf(p[t][r] - mnew);
                p[t][r] = ev;
                rsum += ev;
            }
        rsum += __shfl_xor(rsum, 16, 32);
        lsum = lsum * alpha + rsum;
        mrow = mnew;

        // ---- pack P into 16x32 bf16 A operand (lane-local) ----
        V16BF pa;
#pragma unroll
        for (int j = 0; j < 8; ++j) {
            int t = j >> 2, u = j & 3;
            pa.u[j] = pack2(p[t][2 * u], p[t][2 * u + 1]);
        }

        // ---- per-row alpha broadcast: O row r+8h lives in lane r+8h ----
        float arow[8];
#pragma unroll
        for (int r = 0; r < 8; ++r) arow[r] = __shfl(alpha, r + 8 * h, 32);

        // ---- O[echunk] = alpha*O + P @ V from LDS V^T tile ----
#pragma unroll
        for (int e = 0; e < 32; ++e) {
            const char* vrow = vbuf + (size_t)(e * 16 + m) * VROWB + h * 16;
            V16BF vb;
            vb.u4[0] = ((const uint4*)vrow)[0];
            vb.u4[1] = ((const uint4*)vrow)[2];
            v8f c = o[e];
#pragma unroll
            for (int r = 0; r < 8; ++r) c[r] *= arow[r];
            o[e] = __builtin_amdgcn_wmma_f32_16x16x32_bf16(
                false, pa.v, false, vb.v, (short)0, c, false, false);
        }
        __syncthreads();   // all waves done reading buf before it is refilled
    }

    // ---- finalize: divide by row sums and store ----
    float inv = 1.0f / lsum;
    float irow[8];
#pragma unroll
    for (int r = 0; r < 8; ++r) irow[r] = __shfl(inv, r + 8 * h, 32);
    float* obase = O + ((size_t)b * NL + q0) * NE;
#pragma unroll
    for (int e = 0; e < 32; ++e)
#pragma unroll
        for (int r = 0; r < 8; ++r)
            obase[(size_t)(r + 8 * h) * NE + e * 16 + m] = o[e][r] * irow[r];
}

// -----------------------------------------------------------------------------
extern "C" void kernel_launch(void* const* d_in, const int* in_sizes, int n_in,
                              void* d_out, int out_size, void* d_ws, size_t ws_size,
                              hipStream_t stream) {
    const float* Q = (const float*)d_in[0];
    const float* K = (const float*)d_in[1];
    const float* V = (const float*)d_in[2];
    // d_in[3] = attn_mask: causal triu(k=1); implemented analytically in-kernel.
    float* out = (float*)d_out;

    // workspace: K bf16 (32 MiB) then V^T bf16 (32 MiB)
    uint16_t* kb = (uint16_t*)d_ws;
    uint16_t* vt = kb + (size_t)NB * NS * NE;

    const size_t nK = (size_t)NB * NS * NE;   // 16M elements
    convert_k_kernel<<<dim3((unsigned)(nK / (8 * 256))), dim3(256), 0, stream>>>(K, kb);
    transpose_v_kernel<<<dim3(NS / 32, NE / 32, NB), dim3(32, 8), 0, stream>>>(V, vt);
    flash_attn_kernel<<<dim3(NL / 64, NB), dim3(128), SMEMB, stream>>>(Q, kb, vt, out);
}
